// HedgehogFeatureMap_52063593562939
// MI455X (gfx1250) — matrix operand order repaired and checked by reading, not compiled, not run on hardware
//
#include <hip/hip_runtime.h>

typedef __attribute__((ext_vector_type(16))) _Float16 v16h;
typedef __attribute__((ext_vector_type(8)))  _Float16 v8h;
typedef __attribute__((ext_vector_type(4)))  _Float16 v4h;
typedef __attribute__((ext_vector_type(8)))  float    v8f;

#define DIM    128
#define TILE_M 128

__global__ __launch_bounds__(256)
void hedgehog_wmma_kernel(const float* __restrict__ x,
                          const float* __restrict__ W,
                          const float* __restrict__ b,
                          float* __restrict__ out)
{
    __shared__ __align__(16) _Float16 ldsX[TILE_M * DIM]; // 32 KB f16 x-tile
    __shared__ __align__(16) _Float16 ldsW[DIM * DIM];    // 32 KB f16 W (row n = W[n][0..127])

    const int tid  = threadIdx.x;
    const int wave = tid >> 5;
    const int lane = tid & 31;
    const long long rowBase = (long long)blockIdx.x * TILE_M;

    // ---- Stage x-tile and W into LDS as f16 (float4 loads, packed converts) ----
    {
        const float4* xv = (const float4*)(x + rowBase * DIM);
        const float4* wv = (const float4*)W;
        v4h* lx = (v4h*)ldsX;
        v4h* lw = (v4h*)ldsW;
        #pragma unroll
        for (int i = 0; i < 16; ++i) {
            const int idx = i * 256 + tid;             // 4096 float4 per matrix
            float4 fx = xv[idx];
            float4 fw = wv[idx];
            v4h hx = { (_Float16)fx.x, (_Float16)fx.y, (_Float16)fx.z, (_Float16)fx.w };
            v4h hw = { (_Float16)fw.x, (_Float16)fw.y, (_Float16)fw.z, (_Float16)fw.w };
            lx[idx] = hx;
            lw[idx] = hw;
        }
    }
    __syncthreads();

    const int nm      = lane & 15;   // A: row-in-16; B/C: col-in-16
    const int halfSel = lane >> 4;   // 0 or 1 (lane half)

    // ---- A fragments per ISA 16-bit A layout ----
    // lanes 0-15: K = {0..7, 16..23}; lanes 16-31: K = {8..15, 24..31} (relative to K0)
    v16h aFrag[4];
    #pragma unroll
    for (int kb = 0; kb < 4; ++kb) {
        const int K0   = kb * 32;
        const int koff = halfSel * 8;
        const _Float16* base = &ldsX[(wave * 16 + nm) * DIM + K0 + koff];
        v8h lo = *(const v8h*)(base);        // K0+koff   .. +7
        v8h hi = *(const v8h*)(base + 16);   // K0+16+koff .. +7
        aFrag[kb] = __builtin_shufflevector(lo, hi,
            0,1,2,3,4,5,6,7, 8,9,10,11,12,13,14,15);
    }

    // ---- GEMM: 8 N-tiles x 4 K-steps of v_wmma_f32_16x16x32_f16 ----
    v8f acc[8];
    #pragma unroll
    for (int j = 0; j < 8; ++j)
        #pragma unroll
        for (int e = 0; e < 8; ++e) acc[j][e] = 0.0f;

    #pragma unroll
    for (int j = 0; j < 8; ++j) {
        const int n = j * 16 + nm;           // B column = output col within tile j
        #pragma unroll
        for (int kb = 0; kb < 4; ++kb) {
            // B[k][n] = W[n][k]; same K order as A: {8h..8h+7, 16+8h..16+8h+7}
            const _Float16* bb = &ldsW[n * DIM + kb * 32 + halfSel * 8];
            const v16h bFrag = __builtin_shufflevector(*(const v8h*)(bb), *(const v8h*)(bb + 16),
                0,1,2,3,4,5,6,7, 8,9,10,11,12,13,14,15);
            acc[j] = __builtin_amdgcn_wmma_f32_16x16x32_f16(
                false, aFrag[kb], false, bFrag, (short)0, acc[j], false, false);
        }
    }

    // ---- bias (zeros in reference, but honor it) ----
    float bv[8];
    #pragma unroll
    for (int j = 0; j < 8; ++j) bv[j] = b[j * 16 + nm];

    // ---- dual softmax directly on the C-layout, store ----
    // VGPR a of lane L holds (M = a + 8*halfSel, N = L%16) of each 16x16 tile.
    #pragma unroll
    for (int a = 0; a < 8; ++a) {
        float y[8];
        #pragma unroll
        for (int j = 0; j < 8; ++j) y[j] = acc[j][a] + bv[j];

        float mx = y[0], mn = y[0];
        #pragma unroll
        for (int j = 1; j < 8; ++j) { mx = fmaxf(mx, y[j]); mn = fminf(mn, y[j]); }
        #pragma unroll
        for (int off = 1; off <= 8; off <<= 1) {  // reduces within each 16-lane half
            mx = fmaxf(mx, __shfl_xor(mx, off, 32));
            mn = fminf(mn, __shfl_xor(mn, off, 32));
        }

        float ep[8], en[8];
        float sp = 0.0f, sn = 0.0f;
        #pragma unroll
        for (int j = 0; j < 8; ++j) {
            ep[j] = __expf(y[j] - mx);   sp += ep[j];
            en[j] = __expf(mn - y[j]);   sn += en[j];  // softmax(-y): max(-y) = -min(y)
        }
        #pragma unroll
        for (int off = 1; off <= 8; off <<= 1) {
            sp += __shfl_xor(sp, off, 32);
            sn += __shfl_xor(sn, off, 32);
        }
        const float rp = 1.0f / sp;
        const float rn = 1.0f / sn;

        const long long gm = rowBase + wave * 16 + a + 8 * halfSel;
        float* o = out + gm * DIM;
        #pragma unroll
        for (int j = 0; j < 8; ++j)
            o[j * 16 + nm] = 0.5f * (ep[j] * rp + en[j] * rn);
    }
}

extern "C" void kernel_launch(void* const* d_in, const int* in_sizes, int n_in,
                              void* d_out, int out_size, void* d_ws, size_t ws_size,
                              hipStream_t stream) {
    const float* x = (const float*)d_in[0];
    const float* W = (const float*)d_in[1];
    const float* b = (const float*)d_in[2];
    float* out = (float*)d_out;

    const int M = in_sizes[0] / DIM;         // 4*16*4096 = 262144 rows
    dim3 grid(M / TILE_M);                   // 2048 blocks
    hedgehog_wmma_kernel<<<grid, 256, 0, stream>>>(x, W, b, out);
}
